// IntraRNN_40226663694647
// MI455X (gfx1250) — matrix operand order repaired and checked by reading, not compiled, not run on hardware
//
#include <hip/hip_runtime.h>
#include <hip/hip_bf16.h>

// ---------------------------------------------------------------------------
// CDNA5 / gfx1250 implementation.
//   1) wasi = h @ wa                       (1024x512,  K=1024)  WMMA bf16
//   2) hjua = inter @ ua                   (15360x512, K=1024)  WMMA bf16
//   3) energies/softmax/context/gru_in     (per-row fused, wave32 shuffles)
//   4) gi = gru_in @ W_ih^T + b_ih         (1024x3072, K=1536)  WMMA bf16
//      gh = h      @ W_hh^T + b_hh         (1024x3072, K=1024)  WMMA bf16
//   5) GRU gates -> h_new                  (elementwise)
//   6) output = h_new @ W_out^T + b_out    (1024x50000,K=1024)  WMMA bf16
// GEMM: 256x128 block tile, 8 waves x (64x64), BK=32, double-buffered LDS,
// v_perm_b32-packed f32->bf16 conversion, W_out streamed once (L2 holds A).
// ---------------------------------------------------------------------------

typedef __attribute__((ext_vector_type(16))) __bf16          v16bf;
typedef __attribute__((ext_vector_type(16))) unsigned short  v16u;
typedef __attribute__((ext_vector_type(8)))  float           v8f;

// f32 -> bf16, round-to-nearest (ties away): 2 VALU
__device__ __forceinline__ unsigned short f2bf(float f) {
    return (unsigned short)((__float_as_uint(f) + 0x8000u) >> 16);
}

// pack two f32 -> two bf16 in one u32 (low = x, high = y):
// 2x v_add + 1x v_perm_b32 (selects bytes {y3,y2,x3,x2})
__device__ __forceinline__ unsigned int pk2bf(float x, float y) {
#if __has_builtin(__builtin_amdgcn_cvt_pk_bf16_f32)
    typedef __attribute__((ext_vector_type(2))) __bf16 v2bf;
    v2bf p = __builtin_amdgcn_cvt_pk_bf16_f32(x, y);
    return __builtin_bit_cast(unsigned int, p);
#else
    unsigned int ux = __float_as_uint(x) + 0x8000u;
    unsigned int uy = __float_as_uint(y) + 0x8000u;
    return __builtin_amdgcn_perm(uy, ux, 0x07060302u);
#endif
}

#define BM 256
#define BN 128
#define BK 32
#define LW 18   // LDS row stride in u32 (36 shorts = 72B = 18 banks, conflict-free)

// C[M,N] = A[M,K] @ B + bias.  BT=0: B stored (K,N).  BT=1: B stored (N,K) (A@B^T).
// Requires: M % 256 == 0, K % 32 == 0, N % 16 == 0 (N % 128 only when BT==0).
template<int BT>
__global__ __launch_bounds__(256)
void gemm_bf16(const float* __restrict__ A, const float* __restrict__ B,
               const float* __restrict__ bias, float* __restrict__ C,
               int M, int N, int K, int hasBias)
{
    __shared__ unsigned int As32[2][BM * LW];   // 2 x 18 KB
    __shared__ unsigned int Bs32[2][BN * LW];   // 2 x  9 KB

    const int tid   = threadIdx.x;
    const int mTile = blockIdx.x * BM;
    const int nTile = blockIdx.y * BN;
    const int wave  = tid >> 5;
    const int lane  = tid & 31;
    const int lr    = lane & 15;   // row-in-16 / col-in-16
    const int hi    = lane >> 4;   // half-wave select
    const int wm    = (wave & 3) * 64;   // wave M offset in tile
    const int wn    = (wave >> 2) * 64;  // wave N offset in tile

    v8f acc[4][4] = {};

    const int KT = K / BK;

    // ---- staging helper (write one K-slab into buffer `buf`) ----
    auto stage = [&](int kt, int buf) {
        const int k0 = kt * BK;
        unsigned int* Adst = As32[buf];
        unsigned int* Bdst = Bs32[buf];
        {   // A tile: BM x BK  (32 f32 per thread, packed pair stores)
            const float* Ag = A + (long long)mTile * K + k0;
#pragma unroll
            for (int q = 0; q < 8; ++q) {
                int idx = tid * 32 + q * 4;
                int r = idx >> 5, c = idx & 31;
                const float4 v = *(const float4*)(Ag + (long long)r * K + c);
                Adst[r * LW + (c >> 1) + 0] = pk2bf(v.x, v.y);
                Adst[r * LW + (c >> 1) + 1] = pk2bf(v.z, v.w);
            }
        }
        if (BT) {   // B stored (N,K): Bs[n][k], k contiguous
            const float* Bg = B + k0;
#pragma unroll
            for (int q = 0; q < 4; ++q) {
                int idx = tid * 16 + q * 4;
                int n = idx >> 5, c = idx & 31;
                int gn = nTile + n;
                float4 v = {0.f, 0.f, 0.f, 0.f};
                if (gn < N) {
                    v = *(const float4*)(Bg + (long long)gn * K + c);
                    if (k0 + 2 * BK < K)
                        __builtin_prefetch(Bg + (long long)gn * K + c + 2 * BK, 0, 0);
                }
                Bdst[n * LW + (c >> 1) + 0] = pk2bf(v.x, v.y);
                Bdst[n * LW + (c >> 1) + 1] = pk2bf(v.z, v.w);
            }
        } else {    // B stored (K,N): transpose into Bs[n][k]
            unsigned short* Bss = (unsigned short*)Bdst;
#pragma unroll
            for (int q = 0; q < 4; ++q) {
                int idx = tid * 16 + q * 4;
                int k = idx >> 7, n = idx & 127;
                const float4 v = *(const float4*)(B + (long long)(k0 + k) * N + nTile + n);
                Bss[(n + 0) * 2 * LW + k] = f2bf(v.x);
                Bss[(n + 1) * 2 * LW + k] = f2bf(v.y);
                Bss[(n + 2) * 2 * LW + k] = f2bf(v.z);
                Bss[(n + 3) * 2 * LW + k] = f2bf(v.w);
            }
        }
    };

    stage(0, 0);

    for (int kt = 0; kt < KT; ++kt) {
        __syncthreads();
        if (kt + 1 < KT) stage(kt + 1, (kt + 1) & 1);

        const unsigned short* Ass = (const unsigned short*)As32[kt & 1];
        const unsigned short* Bss = (const unsigned short*)Bs32[kt & 1];

        // ---- build fragments (contiguous k = 16*hi + e, same order in A and B) ----
        v16u au[4], bu[4];
#pragma unroll
        for (int i = 0; i < 4; ++i) {
            const unsigned short* row = &Ass[(wm + i * 16 + lr) * 2 * LW];
#pragma unroll
            for (int e = 0; e < 8; ++e) au[i][e]     = row[hi * 16 + e];       // K 0-7 / 16-23
#pragma unroll
            for (int e = 0; e < 8; ++e) au[i][8 + e] = row[hi * 16 + 8 + e];   // K 8-15 / 24-31
        }
#pragma unroll
        for (int j = 0; j < 4; ++j) {
            const unsigned short* row = &Bss[(wn + j * 16 + lr) * 2 * LW + hi * 16];
#pragma unroll
            for (int e = 0; e < 16; ++e) bu[j][e] = row[e];                    // K 0-15 / 16-31
        }
#pragma unroll
        for (int i = 0; i < 4; ++i) {
            v16bf a = __builtin_bit_cast(v16bf, au[i]);
#pragma unroll
            for (int j = 0; j < 4; ++j) {
                v16bf b = __builtin_bit_cast(v16bf, bu[j]);
                acc[i][j] = __builtin_amdgcn_wmma_f32_16x16x32_bf16(
                    false, a, false, b, (short)0, acc[i][j], false, false);
            }
        }
    }

    // ---- store: VGPR r holds (M = r + 8*hi, N = lr) ----
#pragma unroll
    for (int i = 0; i < 4; ++i) {
        int rowBase = mTile + wm + i * 16 + hi * 8;
#pragma unroll
        for (int j = 0; j < 4; ++j) {
            int col = nTile + wn + j * 16 + lr;
            if (nTile + wn + j * 16 < N) {   // N % 16 == 0 -> whole fragment valid
                float bv = hasBias ? bias[col] : 0.f;
#pragma unroll
                for (int r = 0; r < 8; ++r)
                    C[(long long)(rowBase + r) * N + col] = acc[i][j][r] + bv;
            }
        }
    }
}

// -------- attention: energies -> softmax(15) -> context -> gru_in, x --------
__global__ __launch_bounds__(256)
void attn_kernel(const float* __restrict__ wasi, const float* __restrict__ hjua,
                 const float* __restrict__ inter, const float* __restrict__ va,
                 const float* __restrict__ emb, const int* __restrict__ ids,
                 float* __restrict__ gru_in, float* __restrict__ x_out)
{
    const int b = blockIdx.x;
    const int tid = threadIdx.x, wave = tid >> 5, lane = tid & 31;
    __shared__ float sE[16];
    __shared__ float sAttn[16];

    for (int s = wave; s < 15; s += 8) {
        const float* hj = hjua + (long long)(b * 15 + s) * 512;
        const float* wb = wasi + (long long)b * 512;
        float sum = 0.f;
        for (int h = lane; h < 512; h += 32)
            sum += va[h] * tanhf(wb[h] + hj[h]);
        for (int off = 16; off > 0; off >>= 1)
            sum += __shfl_down(sum, off);
        if (lane == 0) sE[s] = sum;
    }
    __syncthreads();
    if (tid == 0) {
        float mx = -1e30f;
        for (int s = 0; s < 15; ++s) mx = fmaxf(mx, sE[s]);
        float den = 0.f, t[15];
        for (int s = 0; s < 15; ++s) { t[s] = __expf(sE[s] - mx); den += t[s]; }
        float inv = 1.f / den;
        for (int s = 0; s < 15; ++s) sAttn[s] = t[s] * inv;
    }
    __syncthreads();

    for (int e = tid; e < 1024; e += 256) {
        float c = 0.f;
#pragma unroll
        for (int s = 0; s < 15; ++s)
            c += sAttn[s] * inter[(long long)(b * 15 + s) * 1024 + e];
        gru_in[(long long)b * 1536 + 512 + e] = c;
    }
    const int id = ids[b];
    for (int h = tid; h < 512; h += 256) {
        float v = emb[(long long)id * 512 + h];
        gru_in[(long long)b * 1536 + h] = v;
        x_out[(long long)b * 512 + h]   = v;
    }
}

// ------------------------------- GRU gates ---------------------------------
__device__ __forceinline__ float sigmoidf(float x) { return 1.f / (1.f + __expf(-x)); }

__global__ __launch_bounds__(256)
void gates_kernel(const float* __restrict__ gi, const float* __restrict__ gh,
                  const float* __restrict__ h, float* __restrict__ hnew_ws,
                  float* __restrict__ hnew_out, float* __restrict__ gru_out)
{
    int idx = blockIdx.x * 256 + threadIdx.x;        // 1024*1024
    int b = idx >> 10, j = idx & 1023;
    const float* gib = gi + (long long)b * 3072;
    const float* ghb = gh + (long long)b * 3072;
    float r = sigmoidf(gib[j]        + ghb[j]);
    float z = sigmoidf(gib[1024 + j] + ghb[1024 + j]);
    float n = tanhf(gib[2048 + j] + r * ghb[2048 + j]);
    float hv = h[idx];
    float hn = (1.f - z) * n + z * hv;
    hnew_ws[idx] = hn;
    hnew_out[idx] = hn;
    gru_out[idx] = hn;
}

// ---------------------------------------------------------------------------
extern "C" void kernel_launch(void* const* d_in, const int* in_sizes, int n_in,
                              void* d_out, int out_size, void* d_ws, size_t ws_size,
                              hipStream_t stream)
{
    const int*   input_ids = (const int*)  d_in[0];
    const float* hidden    = (const float*)d_in[1];   // (1,1024,1024) -> h
    const float* inter     = (const float*)d_in[2];   // (1024,15,1024)
    const float* emb       = (const float*)d_in[3];   // (50000,512)
    const float* wa        = (const float*)d_in[4];   // (1024,512)  (K,N)
    const float* ua        = (const float*)d_in[5];   // (1024,512)  (K,N)
    const float* va        = (const float*)d_in[6];   // (1,512)
    const float* W_ih      = (const float*)d_in[7];   // (3072,1536) (N,K)
    const float* W_hh      = (const float*)d_in[8];   // (3072,1024) (N,K)
    const float* b_ih      = (const float*)d_in[9];
    const float* b_hh      = (const float*)d_in[10];
    const float* W_out     = (const float*)d_in[11];  // (50000,1024) (N,K)
    const float* b_out     = (const float*)d_in[12];

    float* out        = (float*)d_out;
    float* out_output = out;                    // 51,200,000
    float* out_hnew   = out + 51200000;         //  1,048,576
    float* out_x      = out + 52248576;         //    524,288
    float* out_gru    = out + 52772864;         //  1,048,576

    float* ws     = (float*)d_ws;
    float* wasi   = ws;                 //   524,288 f32
    float* hjua   = ws +   524288;      // 7,864,320 f32
    float* gru_in = ws +  8388608;      // 1,572,864 f32
    float* gi     = ws +  9961472;      // 3,145,728 f32
    float* gh     = ws + 13107200;      // 3,145,728 f32
    float* hnew   = ws + 16252928;      // 1,048,576 f32  (total ~69.2 MB)

    dim3 blk(256);

    // 1) wasi = h @ wa
    gemm_bf16<0><<<dim3(1024 / 256, 512 / 128), blk, 0, stream>>>(
        hidden, wa, nullptr, wasi, 1024, 512, 1024, 0);
    // 2) hjua = inter @ ua
    gemm_bf16<0><<<dim3(15360 / 256, 512 / 128), blk, 0, stream>>>(
        inter, ua, nullptr, hjua, 15360, 512, 1024, 0);
    // 3) attention fuse -> gru_in, x
    attn_kernel<<<dim3(1024), blk, 0, stream>>>(
        wasi, hjua, inter, va, emb, input_ids, gru_in, out_x);
    // 4) gi = gru_in @ W_ih^T + b_ih ; gh = h @ W_hh^T + b_hh
    gemm_bf16<1><<<dim3(1024 / 256, 3072 / 128), blk, 0, stream>>>(
        gru_in, W_ih, b_ih, gi, 1024, 3072, 1536, 1);
    gemm_bf16<1><<<dim3(1024 / 256, 3072 / 128), blk, 0, stream>>>(
        hidden, W_hh, b_hh, gh, 1024, 3072, 1024, 1);
    // 5) gates -> h_new (ws + both output slots)
    gates_kernel<<<dim3((1024 * 1024) / 256), blk, 0, stream>>>(
        gi, gh, hidden, hnew, out_hnew, out_gru);
    // 6) output = h_new @ W_out^T + b_out  (grid.x fastest -> 4 M-tiles reuse
    //    each W_out column block from L2; W_out streams DRAM exactly once)
    gemm_bf16<1><<<dim3(1024 / 256, (50000 + 127) / 128), blk, 0, stream>>>(
        hnew, W_out, b_out, out_output, 1024, 50000, 1024, 1);
}
